// MultiHeadAttention_82214263980101
// MI455X (gfx1250) — matrix operand order repaired and checked by reading, not compiled, not run on hardware
//
#include <hip/hip_runtime.h>

// ---------------------------------------------------------------------------
// CDNA5 (gfx1250) multi-head attention: bf16 WMMA, f32 accumulate,
// TDM (tensor_load_to_lds) staged + double-buffered GEMM B tiles.
// ---------------------------------------------------------------------------

typedef __attribute__((ext_vector_type(16))) __bf16    v16bf;
typedef __attribute__((ext_vector_type(8)))  float     v8f;
typedef __attribute__((ext_vector_type(4)))  unsigned  v4u;
typedef __attribute__((ext_vector_type(8)))  unsigned  v8u;

struct __align__(16) U4 { unsigned a, b, c, d; };
union Frag { v16bf v; U4 q[2]; };

__device__ __forceinline__ unsigned short f2bf(float f) {
    // round-to-nearest-even fp32 -> bf16
    unsigned u = __builtin_bit_cast(unsigned, f);
    unsigned r = u + 0x7FFFu + ((u >> 16) & 1u);
    return (unsigned short)(r >> 16);
}

__device__ __forceinline__ v8f vzero8() {
    v8f z = {0.f, 0.f, 0.f, 0.f, 0.f, 0.f, 0.f, 0.f};
    return z;
}

__device__ __forceinline__ v8f wmma_bf16(const Frag& a, const Frag& b, v8f c) {
    return __builtin_amdgcn_wmma_f32_16x16x32_bf16(
        /*neg_a=*/false, a.v, /*neg_b=*/false, b.v,
        /*c_mod=*/(short)0, c, /*reuse_a=*/false, /*reuse_b=*/false);
}

// ---------------------------------------------------------------------------
// TDM: DMA a 2D bf16 tile (tile0 elems contiguous x tile1 rows, row stride
// stride0 elems) from global into LDS, padding each 64B row by 16B so LDS
// rows land on an 80B (40-short) stride. D# per CDNA5 ISA.
// ---------------------------------------------------------------------------
__device__ __forceinline__ void tdm_load_2d(unsigned lds_off, const void* gaddr,
                                            unsigned tile0, unsigned tile1,
                                            unsigned stride0) {
    unsigned long long ga = (unsigned long long)gaddr;
    v4u g0;
    g0[0] = 1u;                                            // count=1, user mode
    g0[1] = lds_off;                                       // LDS byte address
    g0[2] = (unsigned)ga;                                  // global_addr[31:0]
    g0[3] = (unsigned)((ga >> 32) & 0x1FFFFFFu) | (2u << 30);  // [56:32], type=2
    v8u g1;
    g1[0] = (1u << 16)        // data_size = 2 bytes
          | (1u << 20)        // pad_enable
          | (3u << 22)        // pad_interval: 16 DWORDs (64B) between pads
          | (3u << 25);       // pad_amount: 4 DWORDs (16B)
    const unsigned td0 = 0xFFFFFu, td1 = 0xFFFFFu;         // huge: tiles never OOB
    g1[1] = (td0 & 0xFFFFu) << 16;                         // tensor_dim0[15:0]
    g1[2] = (td0 >> 16) | ((td1 & 0xFFFFu) << 16);         // dim0 hi | dim1 lo
    g1[3] = (td1 >> 16) | (tile0 << 16);                   // dim1 hi | tile_dim0
    g1[4] = tile1 & 0xFFFFu;                               // tile_dim1 (tile_dim2=0)
    g1[5] = stride0;                                       // tensor_dim0_stride lo
    g1[6] = 0u;
    g1[7] = 0u;
    asm volatile("tensor_load_to_lds %0, %1" :: "s"(g0), "s"(g1) : "memory");
}

// ---------------------------------------------------------------------------
// One-time pass: Wt[n*K + k] = bf16(W[k*N + n]) for the 4 weight matrices
// (all 1024x1024), so GEMM B tiles are K-contiguous and TDM-copyable.
// ---------------------------------------------------------------------------
__global__ __launch_bounds__(256)
void transpose_bf16_kernel(const float* __restrict__ W0, const float* __restrict__ W1,
                           const float* __restrict__ W2, const float* __restrict__ W3,
                           unsigned short* __restrict__ Wt, int K, int N)
{
    __shared__ float tile[32][33];
    const float* W = (blockIdx.z == 0) ? W0 : (blockIdx.z == 1) ? W1
                   : (blockIdx.z == 2) ? W2 : W3;
    unsigned short* T = Wt + (size_t)blockIdx.z * K * N;
    const int k0 = blockIdx.x * 32, n0 = blockIdx.y * 32;
    const int tx = threadIdx.x & 31, ty = threadIdx.x >> 5;   // ty: 0..7
#pragma unroll
    for (int r = ty; r < 32; r += 8)
        tile[r][tx] = W[(size_t)(k0 + r) * N + n0 + tx];
    __syncthreads();
#pragma unroll
    for (int r = ty; r < 32; r += 8)
        T[(size_t)(n0 + r) * K + k0 + tx] = f2bf(tile[tx][r]);
}

// ---------------------------------------------------------------------------
// Tiled GEMM:  C[M][N] = A[M][K] * W[K][N] + bias[N],  W given as bf16 Wt[N][K]
//   A_BF16: 0 = fp32 A (convert while staging LDS), 1 = bf16 A
//   OUT_MODE: 0 = fp32 row-major, 1 = bf16 row-major,
//             2 = bf16 "V-transposed"  out[(b*H+h)*64+d][2048]
// Block 256 (8 waves), tile 128x128, K-step 32. B tiles double-buffered via TDM.
// ---------------------------------------------------------------------------
template <int A_BF16, int OUT_MODE>
__global__ __launch_bounds__(256)
void gemm_kernel(const void* __restrict__ Ap, const unsigned short* __restrict__ Wt,
                 const float* __restrict__ bias, void* __restrict__ Cp,
                 int M, int N, int K)
{
    __shared__ unsigned short As[128][40];      // 80B rows (16B aligned, conflict-spread)
    __shared__ unsigned short Bs[2][128][40];   // TDM pads 64B rows to 80B

    const int tid  = threadIdx.x;
    const int lane = tid & 31;
    const int wv   = tid >> 5;
    const int lh   = lane >> 4;
    const int ln   = lane & 15;
    const int m0   = blockIdx.y * 128;
    const int n0   = blockIdx.x * 128;
    const int wrow = (wv >> 1) * 32;            // 4 waves along M
    const int wcol = (wv & 1) * 64;             // 2 waves along N

    v8f acc[2][4];
#pragma unroll
    for (int i = 0; i < 2; ++i)
#pragma unroll
        for (int j = 0; j < 4; ++j) acc[i][j] = vzero8();

    const int ar = tid >> 1, acb = (tid & 1) * 16;   // A tile: 128 rows x 32 k

    const unsigned short* WtBlk = Wt + (size_t)n0 * K;
    const int nIter = K >> 5;

    if (wv == 0)   // prologue: DMA first B tile (per-wave op, EXEC ignored)
        tdm_load_2d((unsigned)(size_t)&Bs[0][0][0], WtBlk, 32u, 128u, (unsigned)K);

    for (int it = 0; it < nIter; ++it) {
        const int k0 = it << 5;
        // ---- stage A tile into LDS (bf16, K-contiguous rows) ----
        if (A_BF16) {
            const unsigned short* Ab = (const unsigned short*)Ap;
            const U4* src = (const U4*)(Ab + (size_t)(m0 + ar) * K + k0 + acb);
            *(U4*)&As[ar][acb]     = src[0];
            *(U4*)&As[ar][acb + 8] = src[1];
        } else {
            const float* Af = (const float*)Ap;
            const float4* src = (const float4*)(Af + (size_t)(m0 + ar) * K + k0 + acb);
#pragma unroll
            for (int c = 0; c < 4; ++c) {
                float4 f = src[c];
                unsigned* dst = (unsigned*)&As[ar][acb + 4 * c];
                dst[0] = (unsigned)f2bf(f.x) | ((unsigned)f2bf(f.y) << 16);
                dst[1] = (unsigned)f2bf(f.z) | ((unsigned)f2bf(f.w) << 16);
            }
            if (k0 + 32 < K)   // L2 prefetch of next fp32 A strip
                __builtin_prefetch(Af + (size_t)(m0 + ar) * K + k0 + 32 + acb, 0, 1);
        }
        // ---- TDM pipeline: kick next B tile, wait for current ----
        if (wv == 0) {
            if (it + 1 < nIter) {
                tdm_load_2d((unsigned)(size_t)&Bs[(it + 1) & 1][0][0],
                            WtBlk + k0 + 32, 32u, 128u, (unsigned)K);
                __builtin_amdgcn_s_wait_tensorcnt(1);  // in-order: current tile done
            } else {
                __builtin_amdgcn_s_wait_tensorcnt(0);
            }
        }
        __syncthreads();

        // ---- fragments + 8 WMMAs per wave ----
        Frag af[2], bf[4];
#pragma unroll
        for (int i = 0; i < 2; ++i) {
            const unsigned short* p = &As[wrow + 16 * i + ln][lh * 8];
            af[i].q[0] = *(const U4*)p;          // K = kb..kb+7
            af[i].q[1] = *(const U4*)(p + 16);   // K = kb+16..kb+23
        }
#pragma unroll
        for (int j = 0; j < 4; ++j) {
            const unsigned short* p = &Bs[it & 1][wcol + 16 * j + ln][lh * 8];
            bf[j].q[0] = *(const U4*)p;          // K = kb..kb+7
            bf[j].q[1] = *(const U4*)(p + 16);   // K = kb+16..kb+23
        }
#pragma unroll
        for (int i = 0; i < 2; ++i)
#pragma unroll
            for (int j = 0; j < 4; ++j)
                acc[i][j] = wmma_bf16(af[i], bf[j], acc[i][j]);
        __syncthreads();
    }

    // ---- epilogue: bias add + store ----
#pragma unroll
    for (int i = 0; i < 2; ++i) {
#pragma unroll
        for (int j = 0; j < 4; ++j) {
            const int ncol = n0 + wcol + 16 * j + ln;
            const float bb = bias[ncol];
#pragma unroll
            for (int v = 0; v < 8; ++v) {
                const int mrow = m0 + wrow + 16 * i + v + 8 * lh;  // C layout
                const float val = acc[i][j][v] + bb;
                if (OUT_MODE == 0) {
                    ((float*)Cp)[(size_t)mrow * N + ncol] = val;
                } else if (OUT_MODE == 1) {
                    ((unsigned short*)Cp)[(size_t)mrow * N + ncol] = f2bf(val);
                } else {
                    // V output pre-transposed: out[(b*H+h)*64+d][s]
                    const size_t idx =
                        ((size_t)(mrow >> 11) * 1024 + ncol) * 2048 + (mrow & 2047);
                    ((unsigned short*)Cp)[idx] = f2bf(val);
                }
            }
        }
    }
}

// ---------------------------------------------------------------------------
// Flash attention. One wave per (batch, head, 16-query tile); keys in steps
// of 32. qw/kw: bf16 [token][H][64]; vwT: bf16 [(b*H+h)*64+d][2048].
// Row sums of P computed by an extra WMMA against an all-ones B matrix.
// ---------------------------------------------------------------------------
__global__ __launch_bounds__(128)
void attn_kernel(const unsigned short* __restrict__ qw,
                 const unsigned short* __restrict__ kw,
                 const unsigned short* __restrict__ vwT,
                 const int* __restrict__ vmask,
                 unsigned short* __restrict__ attn_out)
{
    __shared__ unsigned short Ps[4][16][40];   // per-wave P bounce buffer

    const int lane = threadIdx.x & 31;
    const int wv   = threadIdx.x >> 5;
    const int wid  = blockIdx.x * 4 + wv;
    const int b    = wid >> 11;          // 16 heads * 128 q-tiles per batch
    const int rem  = wid & 2047;
    const int h    = rem >> 7;
    const int q0   = (rem & 127) << 4;
    const int lh   = lane >> 4;
    const int ln   = lane & 15;
    const int kbA  = lh * 8;
    const int kbB  = lh * 8;

    // Q fragments (16 rows x 64 d -> two K=32 chunks), loaded once
    Frag aq[2];
    {
        const unsigned short* qrow =
            qw + (((size_t)(b * 2048 + q0 + ln) * 16 + h) << 6);
#pragma unroll
        for (int kc = 0; kc < 2; ++kc) {
            const unsigned short* p = qrow + kc * 32 + kbA;
            aq[kc].q[0] = *(const U4*)p;
            aq[kc].q[1] = *(const U4*)(p + 16);
        }
    }

    Frag fone;                                   // all-ones bf16 B fragment
    {
        U4 o4; o4.a = o4.b = o4.c = o4.d = 0x3F803F80u;
        fone.q[0] = o4; fone.q[1] = o4;
    }

    v8f oacc[4], lacc;
#pragma unroll
    for (int j = 0; j < 4; ++j) oacc[j] = vzero8();
    lacc = vzero8();
    float m_i[8];
#pragma unroll
    for (int v = 0; v < 8; ++v) m_i[v] = -3.0e38f;

    for (int s0 = 0; s0 < 2048; s0 += 32) {
        // ---- S = Q K^T for two 16-key tiles ----
        v8f sa[2];
        sa[0] = vzero8(); sa[1] = vzero8();
#pragma unroll
        for (int t = 0; t < 2; ++t) {
            const unsigned short* krow =
                kw + (((size_t)(b * 2048 + s0 + 16 * t + ln) * 16 + h) << 6);
#pragma unroll
            for (int kc = 0; kc < 2; ++kc) {
                Frag bk;
                const unsigned short* p = krow + kc * 32 + kbB;
                bk.q[0] = *(const U4*)p;
                bk.q[1] = *(const U4*)(p + 16);
                sa[t] = wmma_bf16(aq[kc], bk, sa[t]);
            }
        }

        // ---- scale + additive key mask ----
        const float madd0 = vmask[b * 2048 + s0 + ln]      ? 0.f : -1e12f;
        const float madd1 = vmask[b * 2048 + s0 + 16 + ln] ? 0.f : -1e12f;
        float sv0[8], sv1[8], mx[8];
#pragma unroll
        for (int v = 0; v < 8; ++v) {
            sv0[v] = sa[0][v] * 0.125f + madd0;
            sv1[v] = sa[1][v] * 0.125f + madd1;
            mx[v]  = fmaxf(sv0[v], sv1[v]);
        }
        // row-max over the 16 lanes of each half (rows live per-VGPR)
#pragma unroll
        for (int v = 0; v < 8; ++v)
#pragma unroll
            for (int off = 1; off < 16; off <<= 1)
                mx[v] = fmaxf(mx[v], __shfl_xor(mx[v], off, 32));

        float corr[8];
#pragma unroll
        for (int v = 0; v < 8; ++v) {
            const float mn = fmaxf(m_i[v], mx[v]);
            corr[v] = __expf(m_i[v] - mn);
            m_i[v]  = mn;
            Ps[wv][v + 8 * lh][ln]      = f2bf(__expf(sv0[v] - mn));
            Ps[wv][v + 8 * lh][16 + ln] = f2bf(__expf(sv1[v] - mn));
        }
        // rescale running O and l by exp(m_old - m_new)
#pragma unroll
        for (int j = 0; j < 4; ++j)
#pragma unroll
            for (int v = 0; v < 8; ++v) oacc[j][v] *= corr[v];
#pragma unroll
        for (int v = 0; v < 8; ++v) lacc[v] *= corr[v];

        // LDS is in-order within a wave; fence the compiler across the bounce
        __builtin_amdgcn_wave_barrier();
        Frag ap;
        {
            const unsigned short* p = &Ps[wv][ln][kbA];
            ap.q[0] = *(const U4*)p;
            ap.q[1] = *(const U4*)(p + 16);
        }
        __builtin_amdgcn_wave_barrier();

        // ---- l += P * ones ; O += P * V ----
        lacc = wmma_bf16(ap, fone, lacc);
#pragma unroll
        for (int j = 0; j < 4; ++j) {
            Frag bvf;
            const unsigned short* p =
                vwT + (((size_t)((b * 16 + h) * 64 + j * 16 + ln)) << 11) + s0 + kbB;
            bvf.q[0] = *(const U4*)p;
            bvf.q[1] = *(const U4*)(p + 16);
            oacc[j] = wmma_bf16(ap, bvf, oacc[j]);
        }
    }

    // ---- epilogue: normalize and store bf16 [token][h*64+d] ----
#pragma unroll
    for (int v = 0; v < 8; ++v) {
        const float inv = 1.0f / lacc[v];
        const int token = b * 2048 + q0 + v + 8 * lh;
#pragma unroll
        for (int j = 0; j < 4; ++j) {
            attn_out[(size_t)token * 1024 + h * 64 + j * 16 + ln] =
                f2bf(oacc[j][v] * inv);
        }
    }
}

// ---------------------------------------------------------------------------
extern "C" void kernel_launch(void* const* d_in, const int* in_sizes, int n_in,
                              void* d_out, int out_size, void* d_ws, size_t ws_size,
                              hipStream_t stream)
{
    const float* q     = (const float*)d_in[0];
    const float* k     = (const float*)d_in[1];
    const float* v     = (const float*)d_in[2];
    const int*   vmask = (const int*)  d_in[3];
    const float* Wq    = (const float*)d_in[4];
    const float* bq    = (const float*)d_in[5];
    const float* Wk    = (const float*)d_in[6];
    const float* bk    = (const float*)d_in[7];
    const float* Wv    = (const float*)d_in[8];
    const float* bv    = (const float*)d_in[9];
    const float* Wo    = (const float*)d_in[10];
    const float* bo    = (const float*)d_in[11];

    const int D = 1024, N = 1024, S = 2048;
    const int M = in_sizes[0] / D;     // B*S
    const int B = M / S;

    unsigned short* ws   = (unsigned short*)d_ws;
    const size_t MN      = (size_t)M * N;
    unsigned short* qw   = ws;
    unsigned short* kw   = qw  + MN;
    unsigned short* vwT  = kw  + MN;
    unsigned short* attn = vwT + MN;
    unsigned short* Wt   = attn + MN;          // 4x bf16 W^T, 1M elems each
    const size_t KN      = (size_t)D * N;

    // one-time: bf16 transposed weights (Wq, Wk, Wv, Wo)
    transpose_bf16_kernel<<<dim3(D / 32, N / 32, 4), 256, 0, stream>>>(
        Wq, Wk, Wv, Wo, Wt, D, N);

    dim3 ggrid(N / 128, M / 128), gblk(256);
    gemm_kernel<0, 1><<<ggrid, gblk, 0, stream>>>(q, Wt,          bq, qw,  M, N, D);
    gemm_kernel<0, 1><<<ggrid, gblk, 0, stream>>>(k, Wt + KN,     bk, kw,  M, N, D);
    gemm_kernel<0, 2><<<ggrid, gblk, 0, stream>>>(v, Wt + 2 * KN, bv, vwT, M, N, D);

    attn_kernel<<<dim3(B * 512), dim3(128), 0, stream>>>(qw, kw, vwT, vmask, attn);

    gemm_kernel<1, 0><<<ggrid, gblk, 0, stream>>>(attn, Wt + 3 * KN, bo, d_out, M, N, N);
}
